// LBLbiLm_65197603554106
// MI455X (gfx1250) — matrix operand order repaired and checked by reading, not compiled, not run on hardware
//
#include <hip/hip_runtime.h>
#include <hip/hip_bf16.h>

// ---------------------------------------------------------------------------
// Problem constants (match reference)
// ---------------------------------------------------------------------------
#define DIM     512          // H
#define ODIM    1024         // 2H
#define SEQ     2048         // S
#define BATCH   32           // B
#define WIN     4            // W
#define NLAYERS 2            // L
#define M_TOTAL (BATCH * SEQ)   // 65536 rows
#define MROWS   32           // rows per block (2 M-tiles of 16)
#define LDSPAD  8            // bf16 row padding for LDS bank spread
#define LDSROW  (DIM + LDSPAD)

typedef __attribute__((ext_vector_type(16))) __bf16 bf16x16;
typedef __attribute__((ext_vector_type(8)))  float  floatx8;
typedef int v4i __attribute__((vector_size(16)));

union Frag {
    bf16x16 v;
    uint4   q[2];
};

#if defined(__gfx1250__) && __has_builtin(__builtin_amdgcn_global_load_async_to_lds_b128)
#define HAVE_ASYNC_LDS 1
#else
#define HAVE_ASYNC_LDS 0
#endif

// ---------------------------------------------------------------------------
// Kernel 1: f32 -> bf16 weight conversion
// ---------------------------------------------------------------------------
__global__ void cvt_bf16_kernel(const float* __restrict__ src,
                                __bf16* __restrict__ dst, int n) {
    int i = blockIdx.x * blockDim.x + threadIdx.x;
    if (i < n) dst[i] = (__bf16)src[i];
}

// ---------------------------------------------------------------------------
// Kernel 2: fused left/right weighted window over the padded sequence.
// Writes f32 state (for the highway carry path) AND a bf16 mirror (GEMM A).
// ---------------------------------------------------------------------------
__global__ __launch_bounds__(256) void conv_kernel(
    const float* __restrict__ inp, const float* __restrict__ lpad,
    const float* __restrict__ rpad, const float* __restrict__ lw,
    const float* __restrict__ rw,
    float* __restrict__ xL, float* __restrict__ xR,
    __bf16* __restrict__ xbL, __bf16* __restrict__ xbR)
{
    const int HV = DIM / 4;                                  // 128 float4 per row
    long tid = (long)blockIdx.x * blockDim.x + threadIdx.x;  // [0, M_TOTAL*HV)
    int  hv  = (int)(tid % HV);
    long bs  = tid / HV;                                     // row m = b*S + s
    int  s   = (int)(bs % SEQ);
    int  b   = (int)(bs / SEQ);

    const float4* inp4 = (const float4*)inp;
    const float4* lp4  = (const float4*)lpad;
    const float4* rp4  = (const float4*)rpad;

    float4 aL; aL.x = aL.y = aL.z = aL.w = 0.0f;
    float4 aR; aR.x = aR.y = aR.z = aR.w = 0.0f;

#pragma unroll
    for (int i = 0; i < WIN; ++i) {
        const float wl = lw[i];   // wave-uniform scalar loads
        const float wr = rw[i];
        // left window: padded row t = s + i (never reaches right pad)
        {
            int t = s + i;
            float4 v = (t < WIN)
                ? lp4[(long)t * HV + hv]
                : inp4[((long)b * SEQ + (t - WIN)) * HV + hv];
            aL.x += wl * v.x; aL.y += wl * v.y;
            aL.z += wl * v.z; aL.w += wl * v.w;
        }
        // right window: padded row t = s + W + 1 + i (never reaches left pad)
        {
            int t = s + WIN + 1 + i;
            float4 v = (t < WIN + SEQ)
                ? inp4[((long)b * SEQ + (t - WIN)) * HV + hv]
                : rp4[(long)(t - WIN - SEQ) * HV + hv];
            aR.x += wr * v.x; aR.y += wr * v.y;
            aR.z += wr * v.z; aR.w += wr * v.w;
        }
    }
    ((float4*)xL)[bs * HV + hv] = aL;
    ((float4*)xR)[bs * HV + hv] = aR;

    union { __bf16 h[4]; uint2 u; } pk;
    pk.h[0] = (__bf16)aL.x; pk.h[1] = (__bf16)aL.y;
    pk.h[2] = (__bf16)aL.z; pk.h[3] = (__bf16)aL.w;
    *reinterpret_cast<uint2*>(xbL + bs * DIM + hv * 4) = pk.u;
    pk.h[0] = (__bf16)aR.x; pk.h[1] = (__bf16)aR.y;
    pk.h[2] = (__bf16)aR.z; pk.h[3] = (__bf16)aR.w;
    *reinterpret_cast<uint2*>(xbR + bs * DIM + hv * 4) = pk.u;
}

// ---------------------------------------------------------------------------
// Kernel 3: one highway layer, both sides (gridDim.y), fused GEMM + gate.
//   proj = x @ W_l^T + b_l ; x' = sig(gate)*x + (1-sig)*relu(nonlinear)
//
// Block = 256 threads (8 waves), owns 32 rows (2 M-tiles).
// Phase 1: async-copy 32x512 bf16 A-strip global->LDS (row-padded).
// Phase 2: wave w handles n-tiles {w+8j}; per (k, j) the Bn/Bg fragments
//          feed 4 WMMAs (two A fragments) -> 2x B reuse vs 16-row blocks.
// Phase 3: gating epilogue via C/D lane layout; writes f32 + bf16 state
//          (or d_out on the last layer).
// ---------------------------------------------------------------------------
__global__ __launch_bounds__(256) void gemm_highway_kernel(
    float* __restrict__ xLbuf, float* __restrict__ xRbuf,
    __bf16* __restrict__ xbL, __bf16* __restrict__ xbR,
    const __bf16* __restrict__ WbL, const __bf16* __restrict__ WbR,
    const float* __restrict__ bL, const float* __restrict__ bR,
    float* __restrict__ out, int layer, int last)
{
    const int side = blockIdx.y;
    float*        x    = side ? xRbuf : xLbuf;
    __bf16*       xb   = side ? xbR : xbL;
    const __bf16* Wb   = (side ? WbR : WbL) + (size_t)layer * ODIM * DIM;
    const float*  bias = (side ? bR : bL) + layer * ODIM;

    __shared__ __bf16 As[MROWS][LDSROW];

    const int  tid  = threadIdx.x;
    const int  lane = tid & 31;
    const int  wave = tid >> 5;
    const long m0   = (long)blockIdx.x * MROWS;

    // -------- Phase 1: 32x512 bf16 strip (contiguous 32KB) -> padded LDS
    {
        const char* gsrc  = (const char*)(xb + m0 * DIM);
        char*       lbase = (char*)&As[0][0];
#pragma unroll
        for (int i = 0; i < 8; ++i) {
            int c      = i * 256 + tid;          // 16B chunk id, [0, 2048)
            int row    = c >> 6;                 // 64 chunks per 1KB row
            int within = c & 63;
            char* l = lbase + row * (int)sizeof(As[0]) + within * 16;
#if HAVE_ASYNC_LDS
            __builtin_amdgcn_global_load_async_to_lds_b128(
                (__attribute__((address_space(1))) v4i*)(gsrc + (size_t)c * 16),
                (__attribute__((address_space(3))) v4i*)l, 0, 0);
#else
            *reinterpret_cast<uint4*>(l) =
                *reinterpret_cast<const uint4*>(gsrc + (size_t)c * 16);
#endif
        }
#if HAVE_ASYNC_LDS
        asm volatile("s_wait_asynccnt 0" ::: "memory");
#endif
        __syncthreads();
    }

    // -------- Phase 2: WMMA main loop
    floatx8 accN[2][4], accG[2][4];
#pragma unroll
    for (int t = 0; t < 2; ++t)
#pragma unroll
        for (int j = 0; j < 4; ++j)
#pragma unroll
            for (int r = 0; r < 8; ++r) { accN[t][j][r] = 0.0f; accG[t][j][r] = 0.0f; }

    // 16-bit A layout: lane l holds row (l&15); K octets {d..d+7, d+16..d+23},
// d = 8*(l>>4).  16-bit B layout: lane l holds col (l&15); same K octets
    // {d..d+7, d+16..d+23}.
    const int a_row = lane & 15;
    const int a_k8  = (lane >> 4) * 8;
    const int b_col = lane & 15;
    const int b_k16 = (lane >> 4) * 8;

    for (int k0 = 0; k0 < DIM; k0 += 32) {
        Frag a0, a1;
        const __bf16* ap0 = &As[a_row][k0 + a_k8];
        const __bf16* ap1 = &As[16 + a_row][k0 + a_k8];
        a0.q[0] = *reinterpret_cast<const uint4*>(ap0);
        a0.q[1] = *reinterpret_cast<const uint4*>(ap0 + 16);
        a1.q[0] = *reinterpret_cast<const uint4*>(ap1);
        a1.q[1] = *reinterpret_cast<const uint4*>(ap1 + 16);
#pragma unroll
        for (int j = 0; j < 4; ++j) {
            const int nt = wave + 8 * j;              // n-tile 0..31
            const int oN = nt * 16 + b_col;           // nonlinear output col
            const int oG = DIM + nt * 16 + b_col;     // gate output col
            Frag bn, bg;
            const uint4* pn = reinterpret_cast<const uint4*>(Wb + (size_t)oN * DIM + k0 + b_k16);
            const uint4* pg = reinterpret_cast<const uint4*>(Wb + (size_t)oG * DIM + k0 + b_k16);
            bn.q[0] = pn[0]; bn.q[1] = pn[2];
            bg.q[0] = pg[0]; bg.q[1] = pg[2];
            accN[0][j] = __builtin_amdgcn_wmma_f32_16x16x32_bf16(
                false, a0.v, false, bn.v, (short)0, accN[0][j], false, false);
            accG[0][j] = __builtin_amdgcn_wmma_f32_16x16x32_bf16(
                false, a0.v, false, bg.v, (short)0, accG[0][j], false, false);
            accN[1][j] = __builtin_amdgcn_wmma_f32_16x16x32_bf16(
                false, a1.v, false, bn.v, (short)0, accN[1][j], false, false);
            accG[1][j] = __builtin_amdgcn_wmma_f32_16x16x32_bf16(
                false, a1.v, false, bg.v, (short)0, accG[1][j], false, false);
        }
    }

    // -------- Phase 3: highway gating epilogue
    // C/D layout: elem r of v8f -> row = r + 8*(lane>>4), col = lane&15.
    const int mbase = (lane >> 4) * 8;
    const int ncol  = lane & 15;
#pragma unroll
    for (int t = 0; t < 2; ++t) {
#pragma unroll
        for (int j = 0; j < 4; ++j) {
            const int nt  = wave + 8 * j;
            const int col = nt * 16 + ncol;           // 0..511
            const float bn = bias[col];
            const float bg = bias[DIM + col];
#pragma unroll
            for (int r = 0; r < 8; ++r) {
                const long m  = m0 + t * 16 + mbase + r;
                const float nl = accN[t][j][r] + bn;
                const float gt = accG[t][j][r] + bg;
                const float g  = 1.0f / (1.0f + expf(-gt));
                const float xo = x[m * DIM + col];
                const float nx = g * xo + (1.0f - g) * fmaxf(nl, 0.0f);
                if (last) {
                    out[m * (size_t)ODIM + (size_t)side * DIM + col] = nx;
                } else {
                    x[m * DIM + col]  = nx;     // one owner per element
                    xb[m * DIM + col] = (__bf16)nx;
                }
            }
        }
    }
}

// ---------------------------------------------------------------------------
// Launch
// ---------------------------------------------------------------------------
extern "C" void kernel_launch(void* const* d_in, const int* in_sizes, int n_in,
                              void* d_out, int out_size, void* d_ws, size_t ws_size,
                              hipStream_t stream) {
    const float* inp  = (const float*)d_in[0];   // [B,S,H]
    const float* lpad = (const float*)d_in[1];   // [W,H]
    const float* rpad = (const float*)d_in[2];   // [W,H]
    const float* lw   = (const float*)d_in[3];   // [W]
    const float* rw   = (const float*)d_in[4];   // [W]
    const float* WL   = (const float*)d_in[5];   // [L,2H,H]
    const float* bLb  = (const float*)d_in[6];   // [L,2H]
    const float* WR   = (const float*)d_in[7];   // [L,2H,H]
    const float* bRb  = (const float*)d_in[8];   // [L,2H]
    float* out = (float*)d_out;                  // [B,S,2H]

    char* ws = (char*)d_ws;
    const size_t xbytes  = (size_t)M_TOTAL * DIM * sizeof(float);    // 128 MB
    const size_t xbbytes = (size_t)M_TOTAL * DIM * sizeof(__bf16);   // 64 MB
    float*  xL  = (float*)ws;
    float*  xR  = (float*)(ws + xbytes);
    __bf16* xbL = (__bf16*)(ws + 2 * xbytes);
    __bf16* xbR = (__bf16*)(ws + 2 * xbytes + xbbytes);
    __bf16* WbL = (__bf16*)(ws + 2 * xbytes + 2 * xbbytes);
    __bf16* WbR = WbL + (size_t)NLAYERS * ODIM * DIM;

    // 1) weights f32 -> bf16
    const int nW = NLAYERS * ODIM * DIM;   // 1,048,576 per side
    cvt_bf16_kernel<<<(nW + 255) / 256, 256, 0, stream>>>(WL, WbL, nW);
    cvt_bf16_kernel<<<(nW + 255) / 256, 256, 0, stream>>>(WR, WbR, nW);

    // 2) windowed context sums (f32 + bf16 mirrors)
    const long convThreads = (long)M_TOTAL * (DIM / 4);   // 8,388,608
    conv_kernel<<<(int)(convThreads / 256), 256, 0, stream>>>(
        inp, lpad, rpad, lw, rw, xL, xR, xbL, xbR);

    // 3) highway layers (both sides per launch via gridDim.y)
    dim3 grid(M_TOTAL / MROWS, 2);
    gemm_highway_kernel<<<grid, 256, 0, stream>>>(
        xL, xR, xbL, xbR, WbL, WbR, bLb, bRb, out, /*layer=*/0, /*last=*/0);
    gemm_highway_kernel<<<grid, 256, 0, stream>>>(
        xL, xR, xbL, xbR, WbL, WbR, bLb, bRb, out, /*layer=*/1, /*last=*/1);
}
